// FusedQKVProjection_30416958390317
// MI455X (gfx1250) — matrix operand order repaired and checked by reading, not compiled, not run on hardware
//
#include <hip/hip_runtime.h>
#include <hip/hip_bf16.h>

typedef __attribute__((ext_vector_type(16))) _Float16 v16h;
typedef __attribute__((ext_vector_type(8)))  _Float16 v8h;
typedef __attribute__((ext_vector_type(8)))  float    v8f;
typedef __attribute__((ext_vector_type(8)))  unsigned int v8u;

// Problem constants (match reference)
#define BATCH    32
#define HIDDEN   4096
#define NH       32
#define NKV      8
#define HD       128
#define MAXSEQ   4096
#define Q_DIM    (NH * HD)             // 4096
#define KV_DIM   (NKV * HD)            // 1024
#define NTOT     (Q_DIM + 2 * KV_DIM)  // 6144
#define KSPLIT   8                     // waves per block, each owns K-slice of 512
#define KSLICE   (HIDDEN / KSPLIT)     // 512

// ---------------------------------------------------------------------------
// FP4 (E2M1) decode via v_perm_b32 byte LUT (unscaled; scale folded post-WMMA).
// f16 encodings of magnitudes {0,.5,1,1.5,2,3,4,6} all have low byte 0x00 and
// high byte {00,38,3C,3E,40,42,44,46}.  v_perm selects bytes 0-7 from the
// {src0,src1} pair: src1 = bytes 0-3 (m=0..3), src0 = bytes 4-7 (m=4..7).
// ---------------------------------------------------------------------------
#define MAG_LO 0x3E3C3800u   // m=0..3 -> 0x00,0x38,0x3C,0x3E
#define MAG_HI 0x46444240u   // m=4..7 -> 0x40,0x42,0x44,0x46

__device__ __forceinline__ void fp4_decode_pair(unsigned p0, unsigned p1,
                                                unsigned& d0, unsigned& d1) {
    unsigned u   = (p0 | (p0 << 4)) & 0x0707u;       // bytes [m0lo, m0hi]
    unsigned v   = (p1 | (p1 << 4)) & 0x0707u;       // bytes [m1lo, m1hi]
    unsigned sel = u | (v << 16);                    // [m0lo,m0hi,m1lo,m1hi]
    unsigned mags = __builtin_amdgcn_perm(MAG_HI, MAG_LO, sel);
    unsigned m0 = __builtin_amdgcn_perm(0u, mags, 0x010C000Cu); // [0,b0,0,b1]
    unsigned m1 = __builtin_amdgcn_perm(0u, mags, 0x030C020Cu); // [0,b2,0,b3]
    d0 = m0 | ((p0 << 12) & 0x8000u) | ((p0 << 24) & 0x80000000u);
    d1 = m1 | ((p1 << 12) & 0x8000u) | ((p1 << 24) & 0x80000000u);
}

// ---------------------------------------------------------------------------
// One-shot x: f32 -> f16 convert (x is tiny: 32x4096; then L2-resident)
// ---------------------------------------------------------------------------
__global__ void xcvt_f32_to_f16(const float* __restrict__ x,
                                _Float16* __restrict__ xh, int n) {
    int i = blockIdx.x * blockDim.x + threadIdx.x;
    if (i < n) xh[i] = (_Float16)x[i];
}

// ---------------------------------------------------------------------------
// Fused QKV GEMM:  D[32 x 6144] = x[32 x 4096] @ dequant(W)^T
// Block = one 16-column group, all 32 batch rows, 8-way split-K.
// Each wave: 32x16 tile over a 512-wide K slice -> 16 steps x 2 WMMA sharing
// ONE decoded B operand (decode VALU halved).  C = inline 0; block scale
// applied per-column in f32 on the partial.  f32 partials reduced in LDS.
// ---------------------------------------------------------------------------
__global__ __launch_bounds__(256)
void qkv_fp4_wmma_kernel(const _Float16* __restrict__ xh,
                         const int*   __restrict__ Wq, const float* __restrict__ Sq,
                         const int*   __restrict__ Wk, const float* __restrict__ Sk,
                         const int*   __restrict__ Wv, const float* __restrict__ Sv,
                         const int*   __restrict__ cache_pos,
                         float* __restrict__ outQ,
                         float* __restrict__ outK,
                         float* __restrict__ outV) {
    __shared__ float red[KSPLIT * 32 * 16];          // 16 KB partials

    const int wave  = threadIdx.x >> 5;
    const int lane  = threadIdx.x & 31;
    const int nbase = blockIdx.x * 16;               // 16-column group

    // Select weight matrix; group never straddles a boundary (multiples of 16).
    const int*   Wp;  const float* Sp;  int colbase;  int dest;
    if (nbase < Q_DIM)               { Wp = Wq; Sp = Sq; colbase = nbase;                  dest = 0; }
    else if (nbase < Q_DIM + KV_DIM) { Wp = Wk; Sp = Sk; colbase = nbase - Q_DIM;          dest = 1; }
    else                             { Wp = Wv; Sp = Sv; colbase = nbase - Q_DIM - KV_DIM; dest = 2; }

    const int nloc = lane & 15;       // B: column within tile;  A: row within tile
    const int hi   = lane >> 4;       // 0: low-K half, 1: high-K half (contiguous k order in A and B)

    const int col = colbase + nloc;
    const int*      prow  = Wp + (size_t)col * (HIDDEN / 2);   // int32 per packed byte
    const float*    srow  = Sp + (size_t)col * (HIDDEN / 32);
    const _Float16* arow0 = xh + (size_t)nloc        * HIDDEN; // rows 0-15
    const _Float16* arow1 = xh + (size_t)(16 + nloc) * HIDDEN; // rows 16-31

    const int kbeg = wave * KSLICE;
    const int kend = kbeg + KSLICE;

    v8f acc0 = {};   // rows 0-15
    v8f acc1 = {};   // rows 16-31

    for (int ks0 = kbeg; ks0 < kend; ks0 += 128) {
        // 4 block scales for this 128-wide K panel in one b128 load.
        const float4 scv = *reinterpret_cast<const float4*>(srow + (ks0 >> 5));
        const float scs[4] = { scv.x, scv.y, scv.z, scv.w };

        if (ks0 + 128 < kend)   // pull next weight panel (global_prefetch_b8)
            __builtin_prefetch(prow + ((ks0 + 128) >> 1) + hi * 8, 0, 1);

#pragma unroll
        for (int j = 0; j < 4; ++j) {
            const int ks = ks0 + j * 32;

            // ---- B operand (raw FP4 -> f16, exact): contiguous k per lane half, same order as A.
            // lanes 0-15 hold K=ks+0..15, lanes 16-31 hold K=ks+16..31.
            // Packed byte holds (even k lo nibble, odd k hi nibble) == one f16 pair.
            const int4* pj = reinterpret_cast<const int4*>(prow + (ks >> 1) + hi * 8);
            int4 b0 = pj[0];
            int4 b1 = pj[1];
            unsigned pvs[8] = { (unsigned)b0.x, (unsigned)b0.y, (unsigned)b0.z, (unsigned)b0.w,
                                (unsigned)b1.x, (unsigned)b1.y, (unsigned)b1.z, (unsigned)b1.w };
            v8u Bu;
#pragma unroll
            for (int i = 0; i < 4; ++i) {
                unsigned d0, d1;
                fp4_decode_pair(pvs[2 * i], pvs[2 * i + 1], d0, d1);
                Bu[2 * i]     = d0;
                Bu[2 * i + 1] = d1;
            }
            v16h B = __builtin_bit_cast(v16h, Bu);

            // ---- A operands: contiguous k order (same as B), one per M-group.
            // lanes 0-15: K = ks+0..7 (v0-3), ks+8..15 (v4-7)
            // lanes16-31: K = ks+16..23 (v0-3), ks+24..31 (v4-7)
            const int bk = ks + hi * 16;
            v8h a00 = *reinterpret_cast<const v8h*>(arow0 + bk);
            v8h a01 = *reinterpret_cast<const v8h*>(arow0 + bk + 8);
            v8h a10 = *reinterpret_cast<const v8h*>(arow1 + bk);
            v8h a11 = *reinterpret_cast<const v8h*>(arow1 + bk + 8);
            v16h A0, A1;
#pragma unroll
            for (int i = 0; i < 8; ++i) {
                A0[i] = a00[i]; A0[8 + i] = a01[i];
                A1[i] = a10[i]; A1[8 + i] = a11[i];
            }

            // Two WMMAs share one decoded B; C = inline 0.
            v8f zero0 = {}, zero1 = {};
            v8f part0 = __builtin_amdgcn_wmma_f32_16x16x32_f16(
                false, A0, false, B, (short)0, zero0, false, false);
            v8f part1 = __builtin_amdgcn_wmma_f32_16x16x32_f16(
                false, A1, false, B, (short)0, zero1, false, false);

            const float scj = scs[j];   // lane-uniform: this lane's column scale
#pragma unroll
            for (int r = 0; r < 8; ++r) {
                acc0[r] += part0[r] * scj;   // f32 FMA (v_pk_fma_f32 pairs)
                acc1[r] += part1[r] * scj;
            }
        }
    }

    // ---- Split-K reduction through LDS.
    // Per lane: 16 contiguous floats: [acc0 r0..7 | acc1 r0..7].
    {
        const int base = (wave * 32 + lane) * 16;
#pragma unroll
        for (int r = 0; r < 8; ++r) {
            red[base + r]     = acc0[r];
            red[base + 8 + r] = acc1[r];
        }
    }
    __syncthreads();

    // 512 output elements, 256 threads -> 2 adjacent elements each.
    const int pos = *cache_pos;
    const int t      = threadIdx.x;
    const int e      = 2 * t;
    const int lane_e = e >> 4;        // source lane 0..31
    const int rr     = e & 15;        // even element index within lane
    float s0 = 0.f, s1 = 0.f;
#pragma unroll
    for (int w = 0; w < KSPLIT; ++w) {
        const int o = (w * 32 + lane_e) * 16 + rr;
        s0 += red[o];
        s1 += red[o + 1];
    }
    // Invert the store layout: D layout -> (m, n).
    // lanes 0-15: VGPR r -> (M=r, N=lane); lanes 16-31: VGPR r -> (M=8+r, N=lane-16)
    const int n   = lane_e & 15;
    const int hie = lane_e >> 4;
    const int m0  = ((rr & 8) ? 16 : 0) + hie * 8 + (rr & 7);   // rr even -> pair (m0, m0+1)

#pragma unroll
    for (int q = 0; q < 2; ++q) {
        const int   m = m0 + q;
        const float v = q ? s1 : s0;
        if (dest == 0) {
            outQ[(size_t)m * Q_DIM + (nbase + n)] = v;
        } else {
            const int cc   = colbase + n;
            const int head = cc >> 7;           // /HD
            const int hd   = cc & 127;          // %HD
            float* basep = (dest == 1) ? outK : outV;
            basep[(((size_t)m * MAXSEQ + pos) * NKV + head) * HD + hd] = v;
        }
    }
}

// ---------------------------------------------------------------------------
// Launch
// ---------------------------------------------------------------------------
extern "C" void kernel_launch(void* const* d_in, const int* in_sizes, int n_in,
                              void* d_out, int out_size, void* d_ws, size_t ws_size,
                              hipStream_t stream) {
    const float* x   = (const float*)d_in[0];
    const int*   Wq  = (const int*)  d_in[1];
    const float* Sq  = (const float*)d_in[2];
    const int*   Wk  = (const int*)  d_in[3];
    const float* Sk  = (const float*)d_in[4];
    const int*   Wv  = (const int*)  d_in[5];
    const float* Sv  = (const float*)d_in[6];
    const float* Kc  = (const float*)d_in[7];
    const float* Vc  = (const float*)d_in[8];
    const int*   pos = (const int*)  d_in[9];

    float* out  = (float*)d_out;
    float* outQ = out;                                         // [32, 4096]
    float* outK = outQ + (size_t)BATCH * Q_DIM;                // [32, 4096, 8, 128]
    float* outV = outK + (size_t)BATCH * MAXSEQ * NKV * HD;    // [32, 4096, 8, 128]

    const size_t cache_bytes = (size_t)BATCH * MAXSEQ * NKV * HD * sizeof(float);

    // Output caches = input caches (then the GEMM scatters row `cache_pos`).
    hipMemcpyAsync(outK, Kc, cache_bytes, hipMemcpyDeviceToDevice, stream);
    hipMemcpyAsync(outV, Vc, cache_bytes, hipMemcpyDeviceToDevice, stream);

    // Stage x as f16 in workspace (256 KB) for WMMA A-operand loads.
    _Float16* xh = (_Float16*)d_ws;
    const int nx = BATCH * HIDDEN;
    xcvt_f32_to_f16<<<(nx + 255) / 256, 256, 0, stream>>>(x, xh, nx);

    // One block per 16-column group; 8 waves = 8-way split-K; 3072 waves total.
    qkv_fp4_wmma_kernel<<<NTOT / 16, 256, 0, stream>>>(
        xh, Wq, Sq, Wk, Sk, Wv, Sv, pos, outQ, outK, outV);
}
